// MultiHeadSpatialAttention_18116172054581
// MI455X (gfx1250) — matrix operand order repaired and checked by reading, not compiled, not run on hardware
//
#include <hip/hip_runtime.h>

#define BB 16
#define CC 256
#define HW_ 1024
#define DM 512
#define NH 8
#define DH 64
#define MTOT (BB * HW_) /* 16384 rows (b,s) */

#if __has_builtin(__builtin_amdgcn_global_load_async_to_lds_b128) && \
    __has_builtin(__builtin_amdgcn_s_wait_asynccnt)
#define USE_ASYNC_LDS 1
#else
#define USE_ASYNC_LDS 0
#endif

typedef __attribute__((ext_vector_type(16))) __bf16 v16bf;
typedef __attribute__((ext_vector_type(8)))  __bf16 v8bf;
typedef __attribute__((ext_vector_type(8)))  float  v8f;

static __device__ __forceinline__ __bf16 f2bf(float f) {
  union { float f; unsigned u; } v; v.f = f;
  unsigned r = v.u + 0x7FFFu + ((v.u >> 16) & 1u);
  unsigned short h = (unsigned short)(r >> 16);
  return __builtin_bit_cast(__bf16, h);
}
static __device__ __forceinline__ float bf2f(__bf16 b) {
  unsigned short h = __builtin_bit_cast(unsigned short, b);
  unsigned u = ((unsigned)h) << 16;
  return __builtin_bit_cast(float, u);
}
static __device__ __forceinline__ v8f zero8() {
  v8f z;
#pragma unroll
  for (int i = 0; i < 8; i++) z[i] = 0.f;
  return z;
}
static __device__ __forceinline__ v8f wmma_bf16(v16bf a, v16bf b, v8f c) {
  return __builtin_amdgcn_wmma_f32_16x16x32_bf16(false, a, false, b, (short)0, c,
                                                 false, false);
}

// A fragment: 16(M) x 32(K) bf16 tile, row-major, stride ld (elements).
// ISA layout: lane r=l&15 row M=r; lane-group hi=l>>4 holds K chunks
// [hi*8, hi*8+8) and [16+hi*8, 16+hi*8+8).
static __device__ __forceinline__ v16bf load_fragA(const __bf16* base, int ld) {
  int lane = threadIdx.x & 31;
  int r = lane & 15, hi = lane >> 4;
  const __bf16* p = base + (size_t)r * ld + hi * 8;
  v8bf c0 = *(const v8bf*)(p);
  v8bf c1 = *(const v8bf*)(p + 16);
  v16bf f;
#pragma unroll
  for (int i = 0; i < 8; i++) { f[i] = c0[i]; f[i + 8] = c1[i]; }
  return f;
}

// B fragment: source is BT = N x K row-major (16 N-rows, 32 K), stride ld.
// ISA layout: lane r=l&15 col N=r; lane-group hi holds K chunks
// [hi*8, hi*8+8) and [16+hi*8, 16+hi*8+8).
static __device__ __forceinline__ v16bf load_fragB(const __bf16* base, int ld) {
  int lane = threadIdx.x & 31;
  int r = lane & 15, hi = lane >> 4;
  const __bf16* p = base + (size_t)r * ld + hi * 8;
  v8bf c0 = *(const v8bf*)(p);
  v8bf c1 = *(const v8bf*)(p + 16);
  v16bf f;
#pragma unroll
  for (int i = 0; i < 8; i++) { f[i] = c0[i]; f[i + 8] = c1[i]; }
  return f;
}

#if USE_ASYNC_LDS
typedef __attribute__((ext_vector_type(4))) int v4i;
typedef __attribute__((address_space(1))) v4i as1_v4i;
typedef __attribute__((address_space(3))) v4i as3_v4i;
// Stage a 64x64 bf16 tile (8 KB) from global (row stride gld elements) into
// LDS (row stride 64), spread across 128 threads: 4 x b128 async per thread.
static __device__ __forceinline__ void async_stage_64x64(const __bf16* gbase,
                                                         int gld, __bf16* lbase) {
  int tx = threadIdx.x;
#pragma unroll
  for (int j = 0; j < 4; j++) {
    int chunk = tx + j * 128;       // 512 chunks of 16 bytes
    int row = chunk >> 3;           // 0..63
    int c8 = (chunk & 7) * 8;       // element offset within row
    const __bf16* g = gbase + (size_t)row * gld + c8;
    __bf16* l = lbase + row * 64 + c8;
    __builtin_amdgcn_global_load_async_to_lds_b128(
        (as1_v4i*)(unsigned long long)g,
        (as3_v4i*)(unsigned)(unsigned long long)l, 0, 0);
  }
}
#endif

// ---------------------------------------------------------------- prep kernels
__global__ void prep_x_kernel(const float* __restrict__ x, __bf16* __restrict__ xr) {
  int idx = blockIdx.x * 256 + threadIdx.x;  // over MTOT*CC
  if (idx >= MTOT * CC) return;
  int c = idx & 255;
  int gm = idx >> 8;  // b*1024 + s
  int b = gm >> 10, s = gm & 1023;
  xr[idx] = f2bf(x[((size_t)b * CC + c) * HW_ + s]);
}

__global__ void prep_w_kernel(const float* __restrict__ W, __bf16* __restrict__ WT,
                              int rows, int cols) {
  // W: rows(K) x cols(N) row-major  ->  WT: cols(N) x rows(K) bf16
  int idx = blockIdx.x * 256 + threadIdx.x;
  if (idx >= rows * cols) return;
  int r = idx / cols, c = idx - r * cols;
  WT[(size_t)c * rows + r] = f2bf(W[idx]);
}

// --------------------------------------------------------- generic WMMA GEMM
// C[M x N] = A[M x K] * BT[N x K]^T (+bias[n]) ; block=128 thr (4 waves 2x2),
// wave tile 64x64. act==1: SiLU. If vt_out != nullptr: store transposed into
// per-head layout vt[b*DM*HW + n*HW + s] (contiguous 8 bf16 per lane).
__global__ __launch_bounds__(128) void gemm_kernel(
    const __bf16* __restrict__ A, const __bf16* __restrict__ BT,
    const float* __restrict__ bias, __bf16* __restrict__ out,
    __bf16* __restrict__ vt_out, int K, int ldout, int act) {
  int tx = threadIdx.x;
  int wave = tx >> 5, lane = tx & 31;
  int wr = wave >> 1, wc = wave & 1;
  int m0 = blockIdx.x * 128 + wr * 64;
  int n0 = blockIdx.y * 128 + wc * 64;

  v8f acc[4][4];
#pragma unroll
  for (int mi = 0; mi < 4; mi++)
#pragma unroll
    for (int ni = 0; ni < 4; ni++) acc[mi][ni] = zero8();

  for (int k0 = 0; k0 < K; k0 += 32) {
    if (k0 + 32 < K) {
      __builtin_prefetch(A + (size_t)m0 * K + k0 + 32);
      __builtin_prefetch(BT + (size_t)n0 * K + k0 + 32);
    }
    v16bf af[4], bf_[4];
#pragma unroll
    for (int mi = 0; mi < 4; mi++)
      af[mi] = load_fragA(A + (size_t)(m0 + mi * 16) * K + k0, K);
#pragma unroll
    for (int ni = 0; ni < 4; ni++)
      bf_[ni] = load_fragB(BT + (size_t)(n0 + ni * 16) * K + k0, K);
#pragma unroll
    for (int mi = 0; mi < 4; mi++)
#pragma unroll
      for (int ni = 0; ni < 4; ni++)
        acc[mi][ni] = wmma_bf16(af[mi], bf_[ni], acc[mi][ni]);
  }

  int lr = lane & 15, lh = lane >> 4;
#pragma unroll
  for (int mi = 0; mi < 4; mi++) {
#pragma unroll
    for (int ni = 0; ni < 4; ni++) {
      int n = n0 + ni * 16 + lr;
      float bv = bias ? bias[n] : 0.f;
      int mbase = m0 + mi * 16 + lh * 8;
      if (vt_out) {
        int b = mbase >> 10, sl = mbase & 1023;
        v8bf pk;
#pragma unroll
        for (int i = 0; i < 8; i++) pk[i] = f2bf(acc[mi][ni][i] + bv);
        *(v8bf*)(vt_out + ((size_t)b * DM + n) * HW_ + sl) = pk;
      } else {
#pragma unroll
        for (int i = 0; i < 8; i++) {
          float v = acc[mi][ni][i] + bv;
          if (act == 1) v = v / (1.f + __expf(-v));  // SiLU
          out[(size_t)(mbase + i) * ldout + n] = f2bf(v);
        }
      }
    }
  }
}

// ------------------------------------------------------------------ attention
// One workgroup per (b, h, 64-query tile). 4 waves; wave w owns 16 query rows.
// Two-pass streaming softmax; S (64x64 f32) and P (64x64 bf16) staged in LDS.
// K/V 64x64 blocks are double-buffered in LDS via async global->LDS copies
// (falls back to direct global fragment loads if builtins are unavailable).
static __device__ __forceinline__ void compute_S_block(const v16bf* qf,
                                                       const __bf16* kbase,
                                                       int kld, float* S) {
  int tx = threadIdx.x;
  int wave = tx >> 5, lane = tx & 31;
  int lr = lane & 15, lh = lane >> 4;
  v8f sacc[4];
#pragma unroll
  for (int nt = 0; nt < 4; nt++) sacc[nt] = zero8();
#pragma unroll
  for (int ks = 0; ks < 2; ks++) {
#pragma unroll
    for (int nt = 0; nt < 4; nt++) {
      v16bf kf = load_fragB(kbase + (size_t)(nt * 16) * kld + ks * 32, kld);
      sacc[nt] = wmma_bf16(qf[ks], kf, sacc[nt]);
    }
  }
#pragma unroll
  for (int nt = 0; nt < 4; nt++) {
    int n = nt * 16 + lr;
#pragma unroll
    for (int i = 0; i < 8; i++)
      S[(wave * 16 + lh * 8 + i) * 64 + n] = sacc[nt][i] * 0.125f;  // 1/sqrt(64)
  }
}

__global__ __launch_bounds__(128) void attn_kernel(
    const __bf16* __restrict__ Qb, const __bf16* __restrict__ Kb,
    const __bf16* __restrict__ Vt, __bf16* __restrict__ comb,
    float* __restrict__ attn_part) {
  __shared__ float S[64 * 64];
  __shared__ __bf16 P[64 * 64];
  __shared__ float mrow[64], lrow[64], colsum[HW_];
#if USE_ASYNC_LDS
  __shared__ __bf16 Kt[2][64 * 64];
  __shared__ __bf16 Vl[2][64 * 64];
#endif

  int bh = blockIdx.x;  // 0..127
  int qt = blockIdx.y;  // 0..15
  int b = bh >> 3, h = bh & 7;
  int q0 = qt * 64;
  int tx = threadIdx.x, wave = tx >> 5, lane = tx & 31;
  int lr = lane & 15, lh = lane >> 4;

  const __bf16* Khead = Kb + (size_t)(b)*HW_ * DM + h * DH;
  const __bf16* Vhead = Vt + ((size_t)b * DM + h * DH) * HW_;
  const __bf16* qbase = Qb + (size_t)(b * HW_ + q0 + wave * 16) * DM + h * DH;
  v16bf qf[2];
  qf[0] = load_fragA(qbase, DM);
  qf[1] = load_fragA(qbase + 32, DM);

  for (int i = tx; i < 64; i += 128) { mrow[i] = -1e30f; lrow[i] = 0.f; }
  for (int i = tx; i < HW_; i += 128) colsum[i] = 0.f;
  __syncthreads();

  // ---- pass 1: row max + sum-exp over all 16 key blocks (K only)
#if USE_ASYNC_LDS
  async_stage_64x64(Khead, DM, Kt[0]);
#endif
  for (int kb = 0; kb < 16; kb++) {
#if USE_ASYNC_LDS
    if (kb + 1 < 16) {
      async_stage_64x64(Khead + (size_t)((kb + 1) * 64) * DM, DM, Kt[(kb + 1) & 1]);
      __builtin_amdgcn_s_wait_asynccnt(4);
    } else {
      __builtin_amdgcn_s_wait_asynccnt(0);
    }
    __syncthreads();
    compute_S_block(qf, Kt[kb & 1], 64, S);
#else
    compute_S_block(qf, Khead + (size_t)(kb * 64) * DM, DM, S);
#endif
    __syncthreads();
    if (tx < 64) {
      float m_old = mrow[tx], bm = m_old;
      for (int c = 0; c < 64; c++) bm = fmaxf(bm, S[tx * 64 + c]);
      float l = lrow[tx] * __expf(m_old - bm);
      for (int c = 0; c < 64; c++) l += __expf(S[tx * 64 + c] - bm);
      mrow[tx] = bm; lrow[tx] = l;
    }
    __syncthreads();
  }

  // ---- pass 2: P = exp(S-m)/l ; O += P @ V ; column sums for attn_map
  v8f oacc[4];
#pragma unroll
  for (int nt = 0; nt < 4; nt++) oacc[nt] = zero8();

#if USE_ASYNC_LDS
  async_stage_64x64(Khead, DM, Kt[0]);
  async_stage_64x64(Vhead, HW_, Vl[0]);
#endif
  for (int kb = 0; kb < 16; kb++) {
#if USE_ASYNC_LDS
    if (kb + 1 < 16) {
      async_stage_64x64(Khead + (size_t)((kb + 1) * 64) * DM, DM, Kt[(kb + 1) & 1]);
      async_stage_64x64(Vhead + (kb + 1) * 64, HW_, Vl[(kb + 1) & 1]);
      __builtin_amdgcn_s_wait_asynccnt(8);
    } else {
      __builtin_amdgcn_s_wait_asynccnt(0);
    }
    __syncthreads();
    compute_S_block(qf, Kt[kb & 1], 64, S);
#else
    compute_S_block(qf, Khead + (size_t)(kb * 64) * DM, DM, S);
#endif
    __syncthreads();
    if (tx < 64) {
      float m = mrow[tx], li = 1.f / lrow[tx];
      for (int c = 0; c < 64; c++)
        P[tx * 64 + c] = f2bf(__expf(S[tx * 64 + c] - m) * li);
    }
    __syncthreads();
    if (tx < 64) {  // column sums of this P block (deterministic, no atomics)
      float s = 0.f;
      for (int r = 0; r < 64; r++) s += bf2f(P[r * 64 + tx]);
      colsum[kb * 64 + tx] += s;
    }
#pragma unroll
    for (int ks = 0; ks < 2; ks++) {
      v16bf pf = load_fragA((const __bf16*)P + (wave * 16) * 64 + ks * 32, 64);
#pragma unroll
      for (int nt = 0; nt < 4; nt++) {
#if USE_ASYNC_LDS
        v16bf vf = load_fragB((const __bf16*)Vl[kb & 1] + (nt * 16) * 64 + ks * 32, 64);
#else
        v16bf vf = load_fragB(Vhead + (size_t)(nt * 16) * HW_ + kb * 64 + ks * 32, HW_);
#endif
        oacc[nt] = wmma_bf16(pf, vf, oacc[nt]);
      }
    }
    __syncthreads();
  }

  // store O tile into combined (b, s, h*64+d) bf16
#pragma unroll
  for (int nt = 0; nt < 4; nt++) {
#pragma unroll
    for (int i = 0; i < 8; i++) {
      int q = q0 + wave * 16 + lh * 8 + i;
      comb[(size_t)(b * HW_ + q) * DM + h * DH + nt * 16 + lr] = f2bf(oacc[nt][i]);
    }
  }
  float* part = attn_part + ((size_t)(bh * 16 + qt)) * HW_;
  for (int i = tx; i < HW_; i += 128) part[i] = colsum[i];
}

__global__ void attn_reduce_kernel(const float* __restrict__ part,
                                   float* __restrict__ out_attn) {
  int idx = blockIdx.x * 256 + threadIdx.x;  // over BB*HW_
  if (idx >= BB * HW_) return;
  int b = idx >> 10, s = idx & 1023;
  float acc = 0.f;
  for (int h = 0; h < NH; h++)
    for (int qt = 0; qt < 16; qt++)
      acc += part[((size_t)((b * NH + h) * 16 + qt)) * HW_ + s];
  out_attn[idx] = acc * (1.f / NH);
}

// -------------------------------------------- fused GEMM + residual + LayerNorm
// 32-row tile, full N=256 per workgroup (so LN reduces in LDS). 4 waves 2x2,
// wave tile 16x128. resid_is_x: residual gathered from NCHW x; else row-major.
__global__ __launch_bounds__(128) void gemm_ln_kernel(
    const __bf16* __restrict__ A, const __bf16* __restrict__ BT,
    const float* __restrict__ bias, const float* __restrict__ resid,
    int resid_is_x, const float* __restrict__ gamma,
    const float* __restrict__ beta, float* __restrict__ out_f32,
    __bf16* __restrict__ out_b16, float* __restrict__ out_chw, int K) {
  __shared__ float T[32 * 256];
  __shared__ float red[32 * 8];

  int tx = threadIdx.x, wave = tx >> 5, lane = tx & 31;
  int wr = wave >> 1, wc = wave & 1;
  int gm0 = blockIdx.x * 32;
  int m0 = wr * 16;
  int n0 = wc * 128;

  v8f acc[8];
#pragma unroll
  for (int ni = 0; ni < 8; ni++) acc[ni] = zero8();

  for (int k0 = 0; k0 < K; k0 += 32) {
    v16bf af = load_fragA(A + (size_t)(gm0 + m0) * K + k0, K);
#pragma unroll
    for (int ni = 0; ni < 8; ni++) {
      v16bf bf_ = load_fragB(BT + (size_t)(n0 + ni * 16) * K + k0, K);
      acc[ni] = wmma_bf16(af, bf_, acc[ni]);
    }
  }

  int lr = lane & 15, lh = lane >> 4;
#pragma unroll
  for (int ni = 0; ni < 8; ni++) {
    int n = n0 + ni * 16 + lr;
    float bv = bias[n];
#pragma unroll
    for (int i = 0; i < 8; i++) {
      int ml = m0 + lh * 8 + i;
      int gm = gm0 + ml;
      float r;
      if (resid_is_x) {
        int b = gm >> 10, s = gm & 1023;
        r = resid[((size_t)b * CC + n) * HW_ + s];
      } else {
        r = resid[(size_t)gm * CC + n];
      }
      T[ml * 256 + n] = acc[ni][i] + bv + r;
    }
  }
  __syncthreads();

  int row = tx >> 2, qq = tx & 3;
  {
    float s = 0.f, s2 = 0.f;
    for (int c = qq * 64; c < qq * 64 + 64; c++) {
      float v = T[row * 256 + c];
      s += v; s2 += v * v;
    }
    red[row * 8 + qq] = s;
    red[row * 8 + 4 + qq] = s2;
  }
  __syncthreads();

  float s = red[row * 8 + 0] + red[row * 8 + 1] + red[row * 8 + 2] + red[row * 8 + 3];
  float s2 = red[row * 8 + 4] + red[row * 8 + 5] + red[row * 8 + 6] + red[row * 8 + 7];
  float mu = s * (1.f / 256.f);
  float var = s2 * (1.f / 256.f) - mu * mu;
  float rs = rsqrtf(var + 1e-5f);
  int gm = gm0 + row;
  for (int c = qq * 64; c < qq * 64 + 64; c++) {
    float v = (T[row * 256 + c] - mu) * rs * gamma[c] + beta[c];
    if (out_f32) out_f32[(size_t)gm * CC + c] = v;
    if (out_b16) out_b16[(size_t)gm * CC + c] = f2bf(v);
    if (out_chw) {
      int b = gm >> 10, sdx = gm & 1023;
      out_chw[((size_t)b * CC + c) * HW_ + sdx] = v;
    }
  }
}

// -------------------------------------------------------------------- launcher
extern "C" void kernel_launch(void* const* d_in, const int* in_sizes, int n_in,
                              void* d_out, int out_size, void* d_ws, size_t ws_size,
                              hipStream_t stream) {
  (void)in_sizes; (void)n_in; (void)out_size; (void)ws_size;
  const float* x  = (const float*)d_in[0];
  const float* Wq = (const float*)d_in[1];  const float* bq  = (const float*)d_in[2];
  const float* Wk = (const float*)d_in[3];  const float* bk  = (const float*)d_in[4];
  const float* Wv = (const float*)d_in[5];  const float* bv  = (const float*)d_in[6];
  const float* Wo = (const float*)d_in[7];  const float* bo  = (const float*)d_in[8];
  const float* W1 = (const float*)d_in[9];  const float* b1  = (const float*)d_in[10];
  const float* W2 = (const float*)d_in[11]; const float* b2  = (const float*)d_in[12];
  const float* g1 = (const float*)d_in[13]; const float* be1 = (const float*)d_in[14];
  const float* g2 = (const float*)d_in[15]; const float* be2 = (const float*)d_in[16];

  float* out = (float*)d_out;
  float* out_attn = out + (size_t)BB * CC * HW_;

  unsigned char* wsp = (unsigned char*)d_ws;
  size_t off = 0;
  auto carve = [&](size_t bytes) -> void* {
    void* p = wsp + off;
    off += (bytes + 255) & ~(size_t)255;
    return p;
  };
  __bf16* WqT  = (__bf16*)carve((size_t)DM * CC * 2);
  __bf16* WkT  = (__bf16*)carve((size_t)DM * CC * 2);
  __bf16* WvT  = (__bf16*)carve((size_t)DM * CC * 2);
  __bf16* WoT  = (__bf16*)carve((size_t)CC * DM * 2);
  __bf16* W1T  = (__bf16*)carve((size_t)CC * CC * 2);
  __bf16* W2T  = (__bf16*)carve((size_t)CC * CC * 2);
  __bf16* xr   = (__bf16*)carve((size_t)MTOT * CC * 2);
  __bf16* Qb   = (__bf16*)carve((size_t)MTOT * DM * 2);
  __bf16* Kb   = (__bf16*)carve((size_t)MTOT * DM * 2);
  __bf16* Vt   = (__bf16*)carve((size_t)MTOT * DM * 2);
  __bf16* comb = (__bf16*)carve((size_t)MTOT * DM * 2);
  float*  ln1f = (float*) carve((size_t)MTOT * CC * 4);
  __bf16* ln1b = (__bf16*)carve((size_t)MTOT * CC * 2);
  __bf16* hmid = (__bf16*)carve((size_t)MTOT * CC * 2);
  float*  part = (float*) carve((size_t)BB * NH * 16 * HW_ * 4);

  // weight / activation prep
  prep_w_kernel<<<(CC * DM + 255) / 256, 256, 0, stream>>>(Wq, WqT, CC, DM);
  prep_w_kernel<<<(CC * DM + 255) / 256, 256, 0, stream>>>(Wk, WkT, CC, DM);
  prep_w_kernel<<<(CC * DM + 255) / 256, 256, 0, stream>>>(Wv, WvT, CC, DM);
  prep_w_kernel<<<(DM * CC + 255) / 256, 256, 0, stream>>>(Wo, WoT, DM, CC);
  prep_w_kernel<<<(CC * CC + 255) / 256, 256, 0, stream>>>(W1, W1T, CC, CC);
  prep_w_kernel<<<(CC * CC + 255) / 256, 256, 0, stream>>>(W2, W2T, CC, CC);
  prep_x_kernel<<<(MTOT * CC) / 256, 256, 0, stream>>>(x, xr);

  // QKV projections: (16384 x 256) @ (256 x 512)
  dim3 gq(MTOT / 128, DM / 128);
  gemm_kernel<<<gq, 128, 0, stream>>>(xr, WqT, bq, Qb, nullptr, CC, DM, 0);
  gemm_kernel<<<gq, 128, 0, stream>>>(xr, WkT, bk, Kb, nullptr, CC, DM, 0);
  gemm_kernel<<<gq, 128, 0, stream>>>(xr, WvT, bv, nullptr, Vt, CC, DM, 0);

  // attention (per b,h, 64-query tiles) + deterministic attn_map reduction
  attn_kernel<<<dim3(BB * NH, HW_ / 64), 128, 0, stream>>>(Qb, Kb, Vt, comb, part);
  attn_reduce_kernel<<<(BB * HW_ + 255) / 256, 256, 0, stream>>>(part, out_attn);

  // out_mapped = LN(comb @ Wo + bo + xr)
  gemm_ln_kernel<<<MTOT / 32, 128, 0, stream>>>(comb, WoT, bo, x, 1, g1, be1,
                                                ln1f, ln1b, nullptr, DM);
  // hmid = silu(ln1 @ W1 + b1)
  gemm_kernel<<<dim3(MTOT / 128, CC / 128), 128, 0, stream>>>(ln1b, W1T, b1, hmid,
                                                              nullptr, CC, CC, 1);
  // out = LN(hmid @ W2 + b2 + ln1)  -> NCHW fp32 into d_out
  gemm_ln_kernel<<<MTOT / 32, 128, 0, stream>>>(hmid, W2T, b2, ln1f, 0, g2, be2,
                                                nullptr, nullptr, out, CC);
}
